// TransformerDecoderLayer_3229815407334
// MI455X (gfx1250) — compile-verified, matrix operand order checked by reading, not run on hardware
//
#include <hip/hip_runtime.h>
#include <hip/hip_bf16.h>

// ---------------------------------------------------------------------------
// Transformer decoder layer (self-attn + cross-attn + MoE) for gfx1250.
// - All matmuls on v_wmma_f32_16x16x32_bf16 (fp32 -> bf16 in-register, f32 acc)
// - GEMM: 128x64 block tile, K-staged 64-deep in LDS, 8 WMMAs/wave per stage
// - MoE hidden kept bf16; its GEMM A-tiles staged via
//   global_load_async_to_lds_b128 (ASYNCcnt) -- no conversion needed
// - Flash attention in transposed (K-major) layout: softmax is per-lane +
//   one shfl_xor(16); P^T feeds the next WMMA B operand with zero shuffles
// ---------------------------------------------------------------------------

typedef __attribute__((ext_vector_type(16))) __bf16 v16bf;
typedef __attribute__((ext_vector_type(8)))  __bf16 v8bf;
typedef __attribute__((ext_vector_type(8)))  float  v8f;

#define BATCH   2
#define SEQ     2048
#define D_MODEL 1024
#define NHEAD   16
#define HDIM    64
#define FFDIM   4096
#define NEXP    8
#define NTOK    (BATCH*SEQ)

__device__ __forceinline__ __bf16 f2bf(float f) {
  unsigned u = __builtin_bit_cast(unsigned, f);
  unsigned r = u + 0x7FFFu + ((u >> 16) & 1u);   // round-to-nearest-even
  unsigned short h = (unsigned short)(r >> 16);
  return __builtin_bit_cast(__bf16, h);
}

__device__ __forceinline__ float geluf(float x) {
  return 0.5f * x * (1.0f + erff(x * 0.70710678118654752f));
}

// 16-element bf16 fragment: elements 0..7 at p[0..7], 8..15 at p[16..23]
// (lane's two K-halves are 16 elements apart in a K-contiguous LDS row).
__device__ __forceinline__ v16bf ldfrag(const __bf16* p) {
  v8bf lo = *(const v8bf*)p;
  v8bf hi = *(const v8bf*)(p + 16);
  return __builtin_shufflevector(lo, hi, 0,1,2,3,4,5,6,7,8,9,10,11,12,13,14,15);
}

// Async bulk copy global->LDS, 16 bytes per lane, tracked by ASYNCcnt.
__device__ __forceinline__ void async_ld_b128(unsigned lds_off, const void* g) {
  asm volatile("global_load_async_to_lds_b128 %0, %1, off"
               :: "v"(lds_off), "v"((unsigned long long)(size_t)g)
               : "memory");
}
__device__ __forceinline__ void wait_asynccnt0() {
  asm volatile("s_wait_asynccnt 0x0" ::: "memory");
}

// ---------------------------------------------------------------------------
// Generic WMMA GEMM:  C[M,N] = op(A[M,K] * B[K,N] + bias)
//   MODE 0: C fp32 = A*B + bias
//   MODE 1: C bf16 = gelu(A*B + bias)                 (MoE hidden)
//   MODE 2: C fp32 += scale[row*NEXP] * (A*B + bias)  (MoE expert accumulate)
// ABF16: A stored bf16 (MoE hidden buffer) -> async-LDS staging path.
// Block tile 128x64, 8 waves, wave tile 32x32, K staged 64-deep.
// ---------------------------------------------------------------------------
template<int MODE, bool ABF16>
__global__ __launch_bounds__(256) void gemm_wmma(const void* __restrict__ Aptr,
                                                 const float* __restrict__ Bw,
                                                 const float* __restrict__ bias,
                                                 const float* __restrict__ scale,
                                                 void* __restrict__ Cptr,
                                                 int N, int Kd)
{
  const int bm   = blockIdx.x * 128;
  const int bn   = blockIdx.y * 64;
  const int tid  = threadIdx.x;
  const int wave = tid >> 5;
  const int lane = tid & 31;
  const int wm   = (wave & 3) * 32;   // 4 waves along M
  const int wn   = (wave >> 2) * 32;  // 2 waves along N
  const int lm   = lane & 15;
  const int kh   = lane >> 4;

  __shared__ __bf16 As[128][72];      // [m][k]  row stride 144B (16B aligned)
  __shared__ __bf16 Bt[64][72];       // [n][k]  transposed B tile

  v8f acc[2][2] = {};

  const float*  Af = (const float*)Aptr;
  const __bf16* Ab = (const __bf16*)Aptr;

  for (int k0 = 0; k0 < Kd; k0 += 64) {
    __syncthreads();
    if (ABF16) {
      // A tile 128x64 bf16 = 16KB: 4 x b128 async copies per thread
      #pragma unroll
      for (int i = 0; i < 4; ++i) {
        int idx = tid + i * 256;            // 1024 16B chunks
        int r = idx >> 3, c8 = (idx & 7) * 8;
        unsigned lds = (unsigned)(size_t)&As[r][c8];
        async_ld_b128(lds, Ab + (size_t)(bm + r) * Kd + (size_t)(k0 + c8));
      }
    } else {
      // A tile 128x64 fp32 -> bf16 convert (32 elems/thread)
      #pragma unroll
      for (int i = 0; i < 32; ++i) {
        int idx = tid + i * 256;
        int r = idx >> 6, c = idx & 63;
        As[r][c] = f2bf(Af[(size_t)(bm + r) * Kd + (size_t)(k0 + c)]);
      }
    }
    // B tile 64x64 -> transposed into Bt[n][k] (16 elems/thread)
    #pragma unroll
    for (int i = 0; i < 16; ++i) {
      int idx = tid + i * 256;
      int r = idx >> 6, c = idx & 63;
      Bt[c][r] = f2bf(Bw[(size_t)(k0 + r) * N + (size_t)(bn + c)]);
    }
    // prefetch next K stage into cache (global_prefetch_b8)
    if (k0 + 64 < Kd) {
      __builtin_prefetch(Bw + (size_t)(k0 + 64 + (tid >> 6)) * N + (size_t)(bn + (tid & 63)), 0, 1);
      if (!ABF16)
        __builtin_prefetch(Af + (size_t)(bm + (tid >> 1)) * Kd + (size_t)(k0 + 64 + (tid & 1) * 32), 0, 1);
    }
    if (ABF16) wait_asynccnt0();
    __syncthreads();

    #pragma unroll
    for (int ks = 0; ks < 2; ++ks) {
      v16bf afr[2], bfr[2];
      #pragma unroll
      for (int m = 0; m < 2; ++m) afr[m] = ldfrag(&As[wm + m * 16 + lm][ks * 32 + kh * 8]);
      #pragma unroll
      for (int n = 0; n < 2; ++n) bfr[n] = ldfrag(&Bt[wn + n * 16 + lm][ks * 32 + kh * 8]);
      #pragma unroll
      for (int m = 0; m < 2; ++m)
        #pragma unroll
        for (int n = 0; n < 2; ++n)
          acc[m][n] = __builtin_amdgcn_wmma_f32_16x16x32_bf16(
              false, afr[m], false, bfr[n], (short)0, acc[m][n], false, false);
    }
  }

  #pragma unroll
  for (int m = 0; m < 2; ++m) {
    #pragma unroll
    for (int n = 0; n < 2; ++n) {
      int col = bn + wn + n * 16 + lm;
      float bv = bias ? bias[col] : 0.0f;
      #pragma unroll
      for (int r = 0; r < 8; ++r) {
        int row = bm + wm + m * 16 + kh * 8 + r;
        float v = acc[m][n][r] + bv;
        size_t o = (size_t)row * N + col;
        if (MODE == 0)      ((float*)Cptr)[o]  = v;
        else if (MODE == 1) ((__bf16*)Cptr)[o] = f2bf(geluf(v));
        else                ((float*)Cptr)[o] += scale[(size_t)row * NEXP] * v;
      }
    }
  }
}

// ---------------------------------------------------------------------------
// Flash attention, transposed-score formulation.
// Block = 128 threads (4 waves); each wave owns 16 query rows of a 64-row
// q tile for one (batch, head).  Per 32-key chunk:
//   S^T[k,q] = K * Q^T            (4 WMMAs, Q pre-scaled by 1/sqrt(HD))
//   online softmax per lane (column q) + shfl_xor(16) across k-halves
//   O^T[d,q] += V^T * P^T         (4 WMMAs; P^T straight from score accs)
// ---------------------------------------------------------------------------
__global__ __launch_bounds__(128) void attn_flash(const float* __restrict__ Qb,
                                                  const float* __restrict__ Kb,
                                                  const float* __restrict__ Vb,
                                                  float* __restrict__ Ob,
                                                  int Sk, int causal)
{
  const int qblk = blockIdx.x;
  const int b    = blockIdx.y / NHEAD;
  const int h    = blockIdx.y % NHEAD;
  const int tid  = threadIdx.x;
  const int wave = tid >> 5;
  const int lane = tid & 31;
  const int lm   = lane & 15;
  const int kh   = lane >> 4;
  const int qb0  = qblk * 64 + wave * 16;

  __shared__ __bf16 Ks[32][72];   // [k][d]   row stride 144B (16B aligned)
  __shared__ __bf16 Vt[64][40];   // [d][k]   transposed V tile

  // Q^T B-fragments over the two 32-wide d-chunks (loaded once, pre-scaled)
  const float* Qrow = Qb + ((size_t)b * SEQ + qb0 + lm) * D_MODEL + h * HDIM;
  v16bf bq[2];
  #pragma unroll
  for (int c = 0; c < 2; ++c)
    #pragma unroll
    for (int e = 0; e < 8; ++e) {
      bq[c][e]     = f2bf(Qrow[c * 32 + kh * 8 + e] * 0.125f);
      bq[c][e + 8] = f2bf(Qrow[c * 32 + 16 + kh * 8 + e] * 0.125f);
    }

  v8f oacc[4] = {};
  float mrun = -1e30f, lsum = 0.0f;

  const float* Kbase = Kb + (size_t)b * Sk * D_MODEL + h * HDIM;
  const float* Vbase = Vb + (size_t)b * Sk * D_MODEL + h * HDIM;
  const int kend = causal ? (qblk * 64 + 64) : Sk;

  for (int k0 = 0; k0 < kend; k0 += 32) {
    __syncthreads();
    #pragma unroll
    for (int i = 0; i < 16; ++i) {
      int idx = tid + i * 128;
      int r = idx >> 6, c = idx & 63;
      float kf = Kbase[(size_t)(k0 + r) * D_MODEL + c];
      float vf = Vbase[(size_t)(k0 + r) * D_MODEL + c];
      Ks[r][c] = f2bf(kf);
      Vt[c][r] = f2bf(vf);
    }
    if (k0 + 32 < kend) {
      int pr = k0 + 32 + (tid >> 2);
      if (pr >= Sk) pr = Sk - 1;
      __builtin_prefetch(Kbase + (size_t)pr * D_MODEL + (size_t)((tid & 3) * 16), 0, 1);
      __builtin_prefetch(Vbase + (size_t)pr * D_MODEL + (size_t)((tid & 3) * 16), 0, 1);
    }
    __syncthreads();

    if (causal && k0 > qb0 + 15) continue;   // chunk fully above this wave's diagonal

    // S^T tiles: rows k' (two 16-row subtiles), cols q
    v8f sacc[2] = {};
    #pragma unroll
    for (int kb = 0; kb < 2; ++kb)
      #pragma unroll
      for (int c = 0; c < 2; ++c) {
        v16bf ak = ldfrag(&Ks[kb * 16 + lm][c * 32 + kh * 8]);
        sacc[kb] = __builtin_amdgcn_wmma_f32_16x16x32_bf16(
            false, ak, false, bq[c], (short)0, sacc[kb], false, false);
      }

    if (causal) {
      int qg = qb0 + lm;
      #pragma unroll
      for (int kb = 0; kb < 2; ++kb)
        #pragma unroll
        for (int r = 0; r < 8; ++r)
          if (k0 + kb * 16 + kh * 8 + r > qg) sacc[kb][r] = -1e30f;
    }

    // online softmax: lane column = query q; combine the two k-halves
    float mloc = -1e30f;
    #pragma unroll
    for (int kb = 0; kb < 2; ++kb)
      #pragma unroll
      for (int r = 0; r < 8; ++r) mloc = fmaxf(mloc, sacc[kb][r]);
    mloc = fmaxf(mloc, __shfl_xor(mloc, 16, 32));
    float mnew  = fmaxf(mrun, mloc);
    float alpha = __expf(mrun - mnew);

    v16bf pfr;          // P^T B-fragment straight from C-layout accumulators
    float ps = 0.0f;
    #pragma unroll
    for (int r = 0; r < 8; ++r) {
      float p0 = __expf(sacc[0][r] - mnew);
      float p1 = __expf(sacc[1][r] - mnew);
      ps += p0 + p1;
      pfr[r]     = f2bf(p0);
      pfr[r + 8] = f2bf(p1);
    }
    lsum = lsum * alpha + ps;

    #pragma unroll
    for (int db = 0; db < 4; ++db) {
      #pragma unroll
      for (int r = 0; r < 8; ++r) oacc[db][r] = oacc[db][r] * alpha;
      v16bf av = ldfrag(&Vt[db * 16 + lm][kh * 8]);
      oacc[db] = __builtin_amdgcn_wmma_f32_16x16x32_bf16(
          false, av, false, pfr, (short)0, oacc[db], false, false);
    }
    mrun = mnew;
  }

  float lt  = lsum + __shfl_xor(lsum, 16, 32);
  float inv = lt > 0.0f ? 1.0f / lt : 0.0f;
  float* Orow = Ob + ((size_t)b * SEQ + qb0 + lm) * D_MODEL + h * HDIM;
  #pragma unroll
  for (int db = 0; db < 4; ++db)
    #pragma unroll
    for (int r = 0; r < 8; ++r)
      Orow[db * 16 + kh * 8 + r] = oacc[db][r] * inv;
}

// ---------------------------------------------------------------------------
// out[row] = layernorm(X[row] + Y[row]) * g + b      (one block per token)
// ---------------------------------------------------------------------------
__global__ __launch_bounds__(256) void add_layernorm(const float* __restrict__ X,
                                                     const float* __restrict__ Y,
                                                     const float* __restrict__ g,
                                                     const float* __restrict__ bta,
                                                     float* __restrict__ out)
{
  const int row = blockIdx.x;
  const float* xr = X + (size_t)row * D_MODEL;
  const float* yr = Y + (size_t)row * D_MODEL;
  float vals[4], s = 0.0f, s2 = 0.0f;
  #pragma unroll
  for (int j = 0; j < 4; ++j) {
    int i = threadIdx.x + j * 256;
    float v = xr[i] + yr[i];
    vals[j] = v; s += v; s2 += v * v;
  }
  __shared__ float rs[256], rq[256];
  rs[threadIdx.x] = s; rq[threadIdx.x] = s2;
  __syncthreads();
  for (int st = 128; st > 0; st >>= 1) {
    if (threadIdx.x < st) {
      rs[threadIdx.x] += rs[threadIdx.x + st];
      rq[threadIdx.x] += rq[threadIdx.x + st];
    }
    __syncthreads();
  }
  float mean = rs[0] * (1.0f / D_MODEL);
  float var  = rq[0] * (1.0f / D_MODEL) - mean * mean;
  float rstd = rsqrtf(var + 1e-5f);
  #pragma unroll
  for (int j = 0; j < 4; ++j) {
    int i = threadIdx.x + j * 256;
    out[(size_t)row * D_MODEL + i] = (vals[j] - mean) * rstd * g[i] + bta[i];
  }
}

// ---------------------------------------------------------------------------
// Router: logits -> softmax -> top-2 -> softmax over the two probabilities
// (faithful to the reference's softmax(top_k(softmax(logits)))).
// ---------------------------------------------------------------------------
__global__ __launch_bounds__(256) void router_topk(const float* __restrict__ x,
                                                   const float* __restrict__ rw,
                                                   const float* __restrict__ rb,
                                                   float* __restrict__ comb)
{
  const int t = blockIdx.x;
  const float* xr = x + (size_t)t * D_MODEL;
  float acc[NEXP] = {};
  for (int i = threadIdx.x; i < D_MODEL; i += 256) {
    float v = xr[i];
    const float* wrow = rw + (size_t)i * NEXP;
    #pragma unroll
    for (int e = 0; e < NEXP; ++e) acc[e] += v * wrow[e];
  }
  __shared__ float red[256][NEXP];
  #pragma unroll
  for (int e = 0; e < NEXP; ++e) red[threadIdx.x][e] = acc[e];
  __syncthreads();
  for (int st = 128; st > 0; st >>= 1) {
    if (threadIdx.x < st)
      #pragma unroll
      for (int e = 0; e < NEXP; ++e) red[threadIdx.x][e] += red[threadIdx.x + st][e];
    __syncthreads();
  }
  if (threadIdx.x == 0) {
    float p[NEXP], mx = -1e30f;
    #pragma unroll
    for (int e = 0; e < NEXP; ++e) { p[e] = red[0][e] + rb[e]; mx = fmaxf(mx, p[e]); }
    float sum = 0.0f;
    #pragma unroll
    for (int e = 0; e < NEXP; ++e) { p[e] = __expf(p[e] - mx); sum += p[e]; }
    #pragma unroll
    for (int e = 0; e < NEXP; ++e) p[e] /= sum;
    int i1 = 0;
    #pragma unroll
    for (int e = 1; e < NEXP; ++e) if (p[e] > p[i1]) i1 = e;
    int i2 = (i1 == 0) ? 1 : 0;
    #pragma unroll
    for (int e = 0; e < NEXP; ++e) if (e != i1 && p[e] > p[i2]) i2 = e;
    float e1 = __expf(p[i1]), e2 = __expf(p[i2]);
    float den = e1 + e2;
    #pragma unroll
    for (int e = 0; e < NEXP; ++e) comb[(size_t)t * NEXP + e] = 0.0f;
    comb[(size_t)t * NEXP + i1] = e1 / den;
    comb[(size_t)t * NEXP + i2] = e2 / den;
  }
}

__global__ __launch_bounds__(256) void zero_f32(float* __restrict__ p, size_t n)
{
  size_t i = (size_t)blockIdx.x * 256 + threadIdx.x;
  if (i < n) p[i] = 0.0f;
}

// ---------------------------------------------------------------------------
extern "C" void kernel_launch(void* const* d_in, const int* in_sizes, int n_in,
                              void* d_out, int out_size, void* d_ws, size_t ws_size,
                              hipStream_t stream)
{
  (void)in_sizes; (void)n_in; (void)out_size; (void)ws_size;

  const float* x        = (const float*)d_in[0];
  const float* enc      = (const float*)d_in[1];
  const float* sa_wq    = (const float*)d_in[2];
  const float* sa_bq    = (const float*)d_in[3];
  const float* sa_wk    = (const float*)d_in[4];
  const float* sa_bk    = (const float*)d_in[5];
  const float* sa_wv    = (const float*)d_in[6];
  const float* sa_bv    = (const float*)d_in[7];
  const float* sa_wo    = (const float*)d_in[8];
  const float* sa_bo    = (const float*)d_in[9];
  const float* ca_wq    = (const float*)d_in[10];
  const float* ca_bq    = (const float*)d_in[11];
  const float* ca_wk    = (const float*)d_in[12];
  const float* ca_bk    = (const float*)d_in[13];
  const float* ca_wv    = (const float*)d_in[14];
  const float* ca_bv    = (const float*)d_in[15];
  const float* ca_wo    = (const float*)d_in[16];
  const float* ca_bo    = (const float*)d_in[17];
  const float* ln1_g    = (const float*)d_in[18];
  const float* ln1_b    = (const float*)d_in[19];
  const float* ln2_g    = (const float*)d_in[20];
  const float* ln2_b    = (const float*)d_in[21];
  const float* ln3_g    = (const float*)d_in[22];
  const float* ln3_b    = (const float*)d_in[23];
  const float* router_w = (const float*)d_in[24];
  const float* router_b = (const float*)d_in[25];
  const float* e_w1     = (const float*)d_in[26];
  const float* e_b1     = (const float*)d_in[27];
  const float* e_w2     = (const float*)d_in[28];
  const float* e_b2     = (const float*)d_in[29];

  // workspace layout (fp32 unless noted)
  float* ws = (float*)d_ws;
  const size_t n = (size_t)NTOK * D_MODEL;      // 4M elements per buffer
  float* qb   = ws;
  float* kb   = qb  + n;
  float* vb   = kb  + n;
  float* ab   = vb  + n;
  float* pb   = ab  + n;
  float* x1   = pb  + n;
  float* x2   = x1  + n;
  float* moe  = x2  + n;
  float* comb = moe + n;                        // [NTOK, NEXP]
  void*  hbuf = (void*)(comb + (size_t)NTOK * NEXP);  // bf16 [NTOK, FFDIM]

  const dim3 gProj(NTOK / 128, D_MODEL / 64);   // 32 x 16
  const dim3 gFF1 (NTOK / 128, FFDIM   / 64);   // 32 x 64
  const dim3 gAttn(SEQ / 64, BATCH * NHEAD);    // 32 x 32

  // ---- causal self-attention ----
  gemm_wmma<0,false><<<gProj, 256, 0, stream>>>(x, sa_wq, sa_bq, nullptr, qb, D_MODEL, D_MODEL);
  gemm_wmma<0,false><<<gProj, 256, 0, stream>>>(x, sa_wk, sa_bk, nullptr, kb, D_MODEL, D_MODEL);
  gemm_wmma<0,false><<<gProj, 256, 0, stream>>>(x, sa_wv, sa_bv, nullptr, vb, D_MODEL, D_MODEL);
  attn_flash<<<gAttn, 128, 0, stream>>>(qb, kb, vb, ab, SEQ, 1);
  gemm_wmma<0,false><<<gProj, 256, 0, stream>>>(ab, sa_wo, sa_bo, nullptr, pb, D_MODEL, D_MODEL);
  add_layernorm<<<NTOK, 256, 0, stream>>>(x, pb, ln1_g, ln1_b, x1);

  // ---- cross-attention ----
  gemm_wmma<0,false><<<gProj, 256, 0, stream>>>(x1,  ca_wq, ca_bq, nullptr, qb, D_MODEL, D_MODEL);
  gemm_wmma<0,false><<<gProj, 256, 0, stream>>>(enc, ca_wk, ca_bk, nullptr, kb, D_MODEL, D_MODEL);
  gemm_wmma<0,false><<<gProj, 256, 0, stream>>>(enc, ca_wv, ca_bv, nullptr, vb, D_MODEL, D_MODEL);
  attn_flash<<<gAttn, 128, 0, stream>>>(qb, kb, vb, ab, SEQ, 0);
  gemm_wmma<0,false><<<gProj, 256, 0, stream>>>(ab, ca_wo, ca_bo, nullptr, pb, D_MODEL, D_MODEL);
  add_layernorm<<<NTOK, 256, 0, stream>>>(x1, pb, ln2_g, ln2_b, x2);

  // ---- MoE ----
  router_topk<<<NTOK, 256, 0, stream>>>(x2, router_w, router_b, comb);
  zero_f32<<<(unsigned)((n + 255) / 256), 256, 0, stream>>>(moe, n);
  for (int e = 0; e < NEXP; ++e) {
    const float* w1 = e_w1 + (size_t)e * D_MODEL * FFDIM;
    const float* b1 = e_b1 + (size_t)e * FFDIM;
    const float* w2 = e_w2 + (size_t)e * FFDIM * D_MODEL;
    const float* b2 = e_b2 + (size_t)e * D_MODEL;
    gemm_wmma<1,false><<<gFF1,  256, 0, stream>>>(x2,   w1, b1, nullptr,  hbuf, FFDIM,   D_MODEL);
    gemm_wmma<2,true ><<<gProj, 256, 0, stream>>>(hbuf, w2, b2, comb + e, moe,  D_MODEL, FFDIM);
  }
  add_layernorm<<<NTOK, 256, 0, stream>>>(x2, moe, ln3_g, ln3_b, (float*)d_out);
}
